// FlashAttention_27745488733024
// MI455X (gfx1250) — matrix operand order repaired and checked by reading, not compiled, not run on hardware
//
#include <hip/hip_runtime.h>

typedef __bf16 bf16_t;
typedef __attribute__((ext_vector_type(16))) __bf16 v16bf;
typedef __attribute__((ext_vector_type(8)))  __bf16 v8bf;
typedef __attribute__((ext_vector_type(4)))  __bf16 v4bf;
typedef __attribute__((ext_vector_type(8)))  float  v8f;
typedef __attribute__((ext_vector_type(4)))  float  v4f;

#define S_LEN   2048
#define DIM     128
#define BS      256                 // key block size (softmax span)
#define NB      (S_LEN / BS)        // 8 key blocks
#define QROWS   128                 // query rows per workgroup
#define NQB     (S_LEN / QROWS)     // 16 query blocks
#define THREADS 512                 // 16 wave32s = 8 wave pairs x 16 rows

// Dynamic LDS layout (CDNA5 WGP: 320KB):
//   kbuf  : [256][128] bf16 row-major              = 65536 B
//   vbufT : [128][256] bf16 ([d][key])             = 65536 B
//   pscr  : 16 waves x [16 rows][128 keys] bf16    = 65536 B
//           (reused as 8 pairs x [16][128] f32 for final O combine)
//   mbuf  : 8 pairs x 32 f32 (partial row maxes)   =  1024 B
//   sbuf  : 8 pairs x 32 f32 (partial row sums)    =  1024 B
#define LDS_BYTES (65536 * 3 + 2048)

union BF16x16 { v16bf v; v8bf h[2]; };

__global__ void __launch_bounds__(THREADS, 1)
blocked_attn_bf16_wmma(const float* __restrict__ Q, const float* __restrict__ K,
                       const float* __restrict__ V, float* __restrict__ O)
{
    extern __shared__ char smem[];
    bf16_t* kbuf  = (bf16_t*)(smem);
    bf16_t* vbufT = (bf16_t*)(smem + 65536);
    bf16_t* pscr  = (bf16_t*)(smem + 131072);
    float*  mbuf  = (float*)(smem + 196608);
    float*  sbuf  = (float*)(smem + 197632);

    const int tid  = threadIdx.x;
    const int lane = tid & 31;
    const int wv   = tid >> 5;     // wave id 0..15
    const int l16  = lane & 15;
    const int hi   = lane >> 4;    // 0: lanes 0-15, 1: lanes 16-31
    const int pair = wv >> 1;      // wave pair 0..7 (shares 16 query rows)
    const int h2   = wv & 1;       // key half: 0 -> keys 0-127, 1 -> keys 128-255

    const int bh = blockIdx.x >> 4;   // fused batch*head 0..31
    const int iq = blockIdx.x & 15;   // 128-row query block 0..15

    const float scale = 0.08838834764831845f;  // 128^-0.5

    const size_t base = (size_t)bh * S_LEN * DIM;
    const float* Qb = Q + base + (size_t)iq * QROWS * DIM;
    const float* Kb = K + base;
    const float* Vb = V + base;
    float*       Ob = O + base + (size_t)iq * QROWS * DIM;

    // ---- Q A-fragments: rows pair*16 .. +15, full d=0..127 (4 frags) ----
    BF16x16 qa[4];
    {
        const float* qrow = Qb + (size_t)(pair * 16 + l16) * DIM;
        #pragma unroll
        for (int f = 0; f < 4; ++f) {
            const int d0 = 32 * f + 8 * hi;
            v4f a0 = *(const v4f*)(qrow + d0);
            v4f a1 = *(const v4f*)(qrow + d0 + 4);
            v4f b0 = *(const v4f*)(qrow + d0 + 16);
            v4f b1 = *(const v4f*)(qrow + d0 + 20);
            v8bf lo, hiv;
            #pragma unroll
            for (int c = 0; c < 4; ++c) {
                lo[c]      = (bf16_t)a0[c];
                lo[c + 4]  = (bf16_t)a1[c];
                hiv[c]     = (bf16_t)b0[c];
                hiv[c + 4] = (bf16_t)b1[c];
            }
            qa[f].h[0] = lo;
            qa[f].h[1] = hiv;
        }
    }

    v8f oacc[8];   // partial O over this wave's key half, accumulated over all j
    #pragma unroll
    for (int vt = 0; vt < 8; ++vt) oacc[vt] = (v8f){0.f,0.f,0.f,0.f,0.f,0.f,0.f,0.f};

    bf16_t* myp = pscr + wv * (16 * 128);   // per-wave private P scratch
    const int xslot = pair * 32 + h2 * 16;       // my mbuf/sbuf slot base
    const int oslot = pair * 32 + (1 - h2) * 16; // partner's slot base

    for (int j = 0; j < NB; ++j) {
        const float* Kj = Kb + (size_t)j * BS * DIM;
        const float* Vj = Vb + (size_t)j * BS * DIM;

        if (j + 1 < NB) {   // global_prefetch_b8 of next tiles
            __builtin_prefetch(Kb + (size_t)(j + 1) * BS * DIM + tid * 64, 0, 0);
            __builtin_prefetch(Vb + (size_t)(j + 1) * BS * DIM + tid * 64, 0, 0);
        }

        // ---- Cooperative staging: K -> kbuf bf16, V -> vbufT bf16 ----
        #pragma unroll
        for (int it = 0; it < (BS * DIM / 4) / THREADS; ++it) {   // 16 iters
            const int elem = (tid + it * THREADS) * 4;
            v4f kk = *(const v4f*)(Kj + elem);
            v4f vv = *(const v4f*)(Vj + elem);
            v4bf kb;
            #pragma unroll
            for (int c = 0; c < 4; ++c) kb[c] = (bf16_t)kk[c];
            *(v4bf*)(kbuf + elem) = kb;
            const int key = elem >> 7;
            const int d   = elem & 127;
            #pragma unroll
            for (int c = 0; c < 4; ++c)
                vbufT[(size_t)(d + c) * BS + key] = (bf16_t)vv[c];
        }
        __syncthreads();

        // ---- S = Q K^T over this wave's 128 keys: 8 tiles, pipelined ----
        v8f sacc[8];
        {
            BF16x16 bb[2];
            auto loadB = [&](int t, BF16x16& d) {
                const int nt = h2 * 8 + (t >> 2), f = t & 3;
                const bf16_t* kr = kbuf + (size_t)(nt * 16 + l16) * DIM
                                 + 32 * f + 8 * hi;
                d.h[0] = *(const v8bf*)(kr);
                d.h[1] = *(const v8bf*)(kr + 16);
            };
            loadB(0, bb[0]);
            const v8f zero = (v8f){0.f,0.f,0.f,0.f,0.f,0.f,0.f,0.f};
            #pragma unroll
            for (int t = 0; t < 32; ++t) {
                if (t + 1 < 32) loadB(t + 1, bb[(t + 1) & 1]);  // overlap next load
                const int nt = t >> 2, f = t & 3;
                sacc[nt] = __builtin_amdgcn_wmma_f32_16x16x32_bf16(
                               false, qa[f].v, false, bb[t & 1].v, (short)0,
                               (f == 0) ? zero : sacc[nt], false, false);
            }
        }

        // ---- Softmax over full 256 keys: in-wave + cross-wave-pair reduce ----
        float mloc[8], sloc[8];
        #pragma unroll
        for (int r = 0; r < 8; ++r) {                 // phase A: partial max
            float m = -3.402823466e38f;
            #pragma unroll
            for (int nt = 0; nt < 8; ++nt) {
                float x = sacc[nt][r] * scale;
                sacc[nt][r] = x;
                m = fmaxf(m, x);
            }
            #pragma unroll
            for (int mask = 1; mask < 16; mask <<= 1)
                m = fmaxf(m, __shfl_xor(m, mask, 32));
            mloc[r] = m;
            if (l16 == 0) mbuf[xslot + 8 * hi + r] = m;
        }
        __syncthreads();
        #pragma unroll
        for (int r = 0; r < 8; ++r) {                 // phase B: exp + partial sum
            const float m = fmaxf(mloc[r], mbuf[oslot + 8 * hi + r]);
            float s = 0.f;
            #pragma unroll
            for (int nt = 0; nt < 8; ++nt) {
                float e = __expf(sacc[nt][r] - m);
                sacc[nt][r] = e;
                s += e;
            }
            #pragma unroll
            for (int mask = 1; mask < 16; mask <<= 1)
                s += __shfl_xor(s, mask, 32);
            sloc[r] = s;
            if (l16 == 0) sbuf[xslot + 8 * hi + r] = s;
        }
        __syncthreads();
        #pragma unroll
        for (int r = 0; r < 8; ++r) {                 // phase C: normalize
            const float inv = 1.0f / (sloc[r] + sbuf[oslot + 8 * hi + r]);
            #pragma unroll
            for (int nt = 0; nt < 8; ++nt) sacc[nt][r] *= inv;
        }

        // ---- Re-stripe P (C-layout -> row-major bf16) into per-wave scratch.
        // Same-wave LDS ops are in-order on CDNA5: wave_barrier suffices.
        #pragma unroll
        for (int nt = 0; nt < 8; ++nt) {
            const int col = nt * 16 + l16;
            #pragma unroll
            for (int r = 0; r < 8; ++r)
                myp[(r + 8 * hi) * 128 + col] = (bf16_t)sacc[nt][r];
        }
        __builtin_amdgcn_wave_barrier();

        // ---- O_partial += P * V over this wave's 128 keys, pipelined ----
        {
            BF16x16 pf[2], vf[2];
            auto loadP = [&](int ktl, BF16x16& d) {
                const bf16_t* pr = myp + l16 * 128 + ktl * 32 + 8 * hi;
                d.h[0] = *(const v8bf*)(pr);
                d.h[1] = *(const v8bf*)(pr + 16);
            };
            auto loadV = [&](int ktl, int vt, BF16x16& d) {
                const bf16_t* vr = vbufT + (size_t)(vt * 16 + l16) * BS
                                 + h2 * 128 + ktl * 32 + 8 * hi;
                d.h[0] = *(const v8bf*)(vr);
                d.h[1] = *(const v8bf*)(vr + 16);
            };
            loadP(0, pf[0]);
            loadV(0, 0, vf[0]);
            #pragma unroll
            for (int t = 0; t < 32; ++t) {
                const int ktl = t >> 3, vt = t & 7;
                const int tn = t + 1;
                if (tn < 32) {
                    const int ktn = tn >> 3, vtn = tn & 7;
                    if (vtn == 0) loadP(ktn, pf[ktn & 1]);
                    loadV(ktn, vtn, vf[tn & 1]);
                }
                oacc[vt] = __builtin_amdgcn_wmma_f32_16x16x32_bf16(
                               false, pf[ktl & 1].v, false, vf[t & 1].v, (short)0,
                               oacc[vt], false, false);
            }
        }
        __syncthreads();   // all waves done with kbuf/vbufT/pscr before restaging
    }

    // ---- Combine wave-pair partials and write O (f32) ----
    float* obuf = (float*)pscr;                       // reuse: 8 pairs x [16][128]
    float* pob  = obuf + (size_t)pair * 16 * 128;
    if (h2 == 1) {
        #pragma unroll
        for (int vt = 0; vt < 8; ++vt)
            #pragma unroll
            for (int r = 0; r < 8; ++r)
                pob[(r + 8 * hi) * 128 + vt * 16 + l16] = oacc[vt][r];
    }
    __syncthreads();
    if (h2 == 0) {
        #pragma unroll
        for (int vt = 0; vt < 8; ++vt) {
            #pragma unroll
            for (int r = 0; r < 8; ++r) {
                const int row = pair * 16 + r + 8 * hi;
                const int col = vt * 16 + l16;
                Ob[(size_t)row * DIM + col] =
                    oacc[vt][r] + pob[(r + 8 * hi) * 128 + col];
            }
        }
    }
}

extern "C" void kernel_launch(void* const* d_in, const int* in_sizes, int n_in,
                              void* d_out, int out_size, void* d_ws, size_t ws_size,
                              hipStream_t stream) {
    const float* Q = (const float*)d_in[0];
    const float* K = (const float*)d_in[1];
    const float* V = (const float*)d_in[2];
    float*       O = (float*)d_out;

    (void)hipFuncSetAttribute((const void*)blocked_attn_bf16_wmma,
                              hipFuncAttributeMaxDynamicSharedMemorySize,
                              LDS_BYTES);

    const int grid = (2 * 16) * NQB;   // B*H * 128-row query blocks = 512
    blocked_attn_bf16_wmma<<<dim3(grid), dim3(THREADS), LDS_BYTES, stream>>>(Q, K, V, O);
}
